// DeeperSubjectLayers_22454089023726
// MI455X (gfx1250) — matrix operand order repaired and checked by reading, not compiled, not run on hardware
//
#include <hip/hip_runtime.h>

// Problem constants (match the reference)
#define BATCH 128
#define CH    256
#define DIM   256
#define TLEN  2048
#define TN    128   // T columns per workgroup
#define KC    32    // K (channel) step per WMMA (bf16 K=32)

typedef __attribute__((ext_vector_type(16))) __bf16       v16bf;
typedef __attribute__((ext_vector_type(4)))  __bf16       v4bf;
typedef __attribute__((ext_vector_type(8)))  float        v8f;
typedef __attribute__((ext_vector_type(4)))  unsigned int v4u;

union FragU { v4u u[2]; v16bf v; };

// 4x fp32 -> packed 4x bf16 via native converts (v_cvt_pk_bf16_f32)
__device__ __forceinline__ unsigned long long pack4(float a, float b, float c, float d) {
    v4bf v = { (__bf16)a, (__bf16)b, (__bf16)c, (__bf16)d };
    return __builtin_bit_cast(unsigned long long, v);
}

__global__ __launch_bounds__(256)
void subj_gemm_wmma(const float* __restrict__ x,
                    const int*   __restrict__ subj,
                    const float* __restrict__ w,
                    float* __restrict__ out) {
    // LDS tiles, K-contiguous so WMMA fragments are contiguous ds_load_b128s
    __shared__ unsigned short Ws[DIM * KC]; // A-side: Ws[m][k] = w[s][kk+k][m]    (16 KB)
    __shared__ unsigned short Xs[TN  * KC]; // B-side: Xs[t][k] = x[b][kk+k][t0+t] ( 8 KB)

    const int tid  = threadIdx.x;
    const int b    = blockIdx.y;
    const int t0   = blockIdx.x * TN;
    const int s    = subj[b];

    const float* xb = x   + (size_t)b * CH * TLEN;
    const float* ws = w   + (size_t)s * CH * DIM;
    float*       ob = out + (size_t)b * DIM * TLEN;

    const int wave   = tid >> 5;
    const int wave_m = wave >> 1;   // 0..3 : which group of 4 M-tiles
    const int wave_n = wave & 1;    // 0..1 : which group of 4 N-tiles
    const int lane   = tid & 31;
    const int hs     = lane >> 4;   // half-wave select
    const int l16    = lane & 15;

    v8f acc[4][4];
#pragma unroll
    for (int i = 0; i < 4; ++i)
#pragma unroll
        for (int j = 0; j < 4; ++j)
            acc[i][j] = v8f{};

    for (int kk = 0; kk < CH; kk += KC) {
        // ---- stage X tile: 32 (k) x 128 (t) fp32 -> bf16, transposed to [t][k]
        {
            const int tb = tid & 31;   // 32 t-blocks of 4
            const int kb = tid >> 5;   // 8  k-blocks of 4
            const float* src = xb + (size_t)(kk + kb * 4) * TLEN + t0 + tb * 4;
            const float4 r0 = *(const float4*)(src);
            const float4 r1 = *(const float4*)(src + TLEN);
            const float4 r2 = *(const float4*)(src + 2 * TLEN);
            const float4 r3 = *(const float4*)(src + 3 * TLEN);
            const int base = (tb * 4) * KC + kb * 4;
            *(unsigned long long*)&Xs[base         ] = pack4(r0.x, r1.x, r2.x, r3.x);
            *(unsigned long long*)&Xs[base + 1 * KC] = pack4(r0.y, r1.y, r2.y, r3.y);
            *(unsigned long long*)&Xs[base + 2 * KC] = pack4(r0.z, r1.z, r2.z, r3.z);
            *(unsigned long long*)&Xs[base + 3 * KC] = pack4(r0.w, r1.w, r2.w, r3.w);
            // WGP-scope prefetch (locality 3) of next K-step's x rows: pulls into
            // all cache levels -- this WGP consumes them one iteration later.
            if (kk + KC < CH)
                __builtin_prefetch(src + (size_t)KC * TLEN, 0, 3);
        }
        // ---- stage W tile: 32 (k) x 256 (d) fp32 -> bf16, transposed to [m][k]
#pragma unroll
        for (int it = 0; it < 2; ++it) {
            const int i  = tid + it * 256;  // 512 supertiles of 4x4
            const int db = i & 63;          // 64 d-blocks of 4
            const int kb = i >> 6;          // 8  k-blocks of 4
            const float* src = ws + (size_t)(kk + kb * 4) * DIM + db * 4;
            const float4 r0 = *(const float4*)(src);
            const float4 r1 = *(const float4*)(src + DIM);
            const float4 r2 = *(const float4*)(src + 2 * DIM);
            const float4 r3 = *(const float4*)(src + 3 * DIM);
            const int base = (db * 4) * KC + kb * 4;
            *(unsigned long long*)&Ws[base         ] = pack4(r0.x, r1.x, r2.x, r3.x);
            *(unsigned long long*)&Ws[base + 1 * KC] = pack4(r0.y, r1.y, r2.y, r3.y);
            *(unsigned long long*)&Ws[base + 2 * KC] = pack4(r0.z, r1.z, r2.z, r3.z);
            *(unsigned long long*)&Ws[base + 3 * KC] = pack4(r0.w, r1.w, r2.w, r3.w);
        }
        __syncthreads();

        // ---- load fragments per the CDNA5 16-bit WMMA VGPR layouts
        FragU a[4], bq[4];
#pragma unroll
        for (int i = 0; i < 4; ++i) {
            const int arow = (wave_m * 4 + i) * 16 + l16;  // M row
            const int ak   = hs * 8;                       // lanes>=16: K 8..15 / 24..31
            a[i].u[0] = *(const v4u*)&Ws[arow * KC + ak];
            a[i].u[1] = *(const v4u*)&Ws[arow * KC + ak + 16];
        }
#pragma unroll
        for (int j = 0; j < 4; ++j) {
            const int brow = (wave_n * 4 + j) * 16 + l16;  // N column
            const int bk   = hs * 8;                       // lanes>=16: K 8..15 / 24..31
            bq[j].u[0] = *(const v4u*)&Xs[brow * KC + bk];
            bq[j].u[1] = *(const v4u*)&Xs[brow * KC + bk + 16];
        }

        // ---- 16 WMMAs per K-step per wave (1:1 with ds_load_b128)
#pragma unroll
        for (int i = 0; i < 4; ++i)
#pragma unroll
            for (int j = 0; j < 4; ++j)
                acc[i][j] = __builtin_amdgcn_wmma_f32_16x16x32_bf16(
                    false, a[i].v, false, bq[j].v, (short)0, acc[i][j], false, false);

        __syncthreads();
    }

    // ---- epilogue: C/D layout -> out[b][d][t] (16 consecutive lanes per store row)
#pragma unroll
    for (int i = 0; i < 4; ++i) {
#pragma unroll
        for (int j = 0; j < 4; ++j) {
#pragma unroll
            for (int r = 0; r < 8; ++r) {
                const int d = (wave_m * 4 + i) * 16 + r + hs * 8;
                const int t = t0 + (wave_n * 4 + j) * 16 + l16;
                ob[(size_t)d * TLEN + t] = acc[i][j][r];
            }
        }
    }
}

extern "C" void kernel_launch(void* const* d_in, const int* in_sizes, int n_in,
                              void* d_out, int out_size, void* d_ws, size_t ws_size,
                              hipStream_t stream) {
    const float* x        = (const float*)d_in[0];
    const int*   subjects = (const int*)d_in[1];
    const float* weights  = (const float*)d_in[2];
    float*       out      = (float*)d_out;

    dim3 grid(TLEN / TN, BATCH);  // 16 x 128 workgroups
    subj_gemm_wmma<<<grid, 256, 0, stream>>>(x, subjects, weights, out);
}
